// MoE_8796093022942
// MI455X (gfx1250) — matrix operand order repaired and checked by reading, not compiled, not run on hardware
//
#include <hip/hip_runtime.h>

// Problem constants (B,S,D,H,E,TOPK) = (4,2048,1024,2048,8,2)
#define T_TOK   8192
#define DMODEL  1024
#define HDIM    2048
#define NEXP    8
#define MTILE   32         // tokens per tile (two 16-row WMMA tiles sharing B)
#define NSLOT   16640      // T*TOPK (16384) + 8*32 padding headroom
#define MAX_TILES 520      // 16384/32 + 8
#define KCHUNK  64         // K elements staged in LDS per buffer (32 rows x 64 = 16B/thread)

typedef __attribute__((ext_vector_type(16))) __bf16       v16bf;
typedef __attribute__((ext_vector_type(8)))  float        v8f;
typedef __attribute__((ext_vector_type(4)))  float        v4f;
typedef __attribute__((ext_vector_type(4)))  unsigned int v4u;
typedef __attribute__((ext_vector_type(2)))  unsigned int v2u;

static __device__ __forceinline__ unsigned int f2bf1(float f) {
  unsigned int u = __float_as_uint(f);
  return (u + 0x7FFFu + ((u >> 16) & 1u)) >> 16;   // round-to-nearest-even
}
static __device__ __forceinline__ unsigned int f2bf2(float lo, float hi) {
  return f2bf1(lo) | (f2bf1(hi) << 16);
}

union AVec { v16bf v; struct { v4u lo; v4u hi; } q; };

// ---------------- 1) generic f32 -> bf16 cast (pre-pass, bandwidth bound) ------
__global__ __launch_bounds__(256) void cast_bf16_kernel(const float* __restrict__ src,
                                                        unsigned short* __restrict__ dst) {
  int i = blockIdx.x * 256 + threadIdx.x;   // one group of 4 elements per thread
  v4f f = ((const v4f*)src)[i];
  v2u o; o.x = f2bf2(f.x, f.y); o.y = f2bf2(f.z, f.w);
  ((v2u*)dst)[i] = o;
}

// ---------------- 2) router: wave-per-token, f32, top-2 ----------------
__global__ __launch_bounds__(256) void router_kernel(const float* __restrict__ x,
                                                     const float* __restrict__ rw,
                                                     int2* __restrict__ e01,
                                                     float2* __restrict__ w01,
                                                     int* __restrict__ counts) {
  int t    = blockIdx.x * 8 + (threadIdx.x >> 5);
  int lane = threadIdx.x & 31;
  const float* xr = x + (size_t)t * DMODEL;
  float acc[NEXP];
#pragma unroll
  for (int e = 0; e < NEXP; ++e) acc[e] = 0.f;
  for (int k = lane; k < DMODEL; k += 32) {
    float xv = xr[k];
#pragma unroll
    for (int e = 0; e < NEXP; ++e) acc[e] += xv * rw[e * DMODEL + k];
  }
#pragma unroll
  for (int e = 0; e < NEXP; ++e)
    for (int m = 16; m >= 1; m >>= 1) acc[e] += __shfl_xor(acc[e], m, 32);
  if (lane == 0) {
    float mx = acc[0];
#pragma unroll
    for (int e = 1; e < NEXP; ++e) mx = fmaxf(mx, acc[e]);
    float p[NEXP];
#pragma unroll
    for (int e = 0; e < NEXP; ++e) p[e] = __expf(acc[e] - mx);
    int i0 = 0; float b0 = p[0];
#pragma unroll
    for (int e = 1; e < NEXP; ++e) if (p[e] > b0) { b0 = p[e]; i0 = e; }
    int i1 = (i0 == 0) ? 1 : 0; float b1 = p[i1];
#pragma unroll
    for (int e = 0; e < NEXP; ++e) if (e != i0 && p[e] > b1) { b1 = p[e]; i1 = e; }
    float s = b0 + b1;              // softmax denominator cancels in renorm
    e01[t] = make_int2(i0, i1);
    w01[t] = make_float2(b0 / s, b1 / s);
    atomicAdd(&counts[i0], 1);
    atomicAdd(&counts[i1], 1);
  }
}

// ---------------- 3) setup: padded bases, tile table, pad slots ----------------
__global__ void setup_kernel(const int* __restrict__ counts, int* __restrict__ cursor,
                             int* __restrict__ token_list, float* __restrict__ weight_list,
                             int* __restrict__ tiles_e, int* __restrict__ tiles_base,
                             int* __restrict__ ntiles) {
  if (threadIdx.x != 0 || blockIdx.x != 0) return;
  int base = 0, tt = 0;
  for (int e = 0; e < NEXP; ++e) {
    int cnt   = counts[e];
    int tiles = (cnt + MTILE - 1) / MTILE;
    cursor[e] = base;
    for (int s = cnt; s < tiles * MTILE; ++s) { token_list[base + s] = 0; weight_list[base + s] = 0.f; }
    for (int k = 0; k < tiles; ++k) { tiles_e[tt] = e; tiles_base[tt] = base + k * MTILE; ++tt; }
    base += tiles * MTILE;
  }
  *ntiles = tt;
}

// ---------------- 4) scatter tokens into per-expert slot lists ----------------
__global__ __launch_bounds__(256) void scatter_kernel(const int2* __restrict__ e01,
                                                      const float2* __restrict__ w01,
                                                      int* __restrict__ cursor,
                                                      int* __restrict__ token_list,
                                                      float* __restrict__ weight_list) {
  int t = blockIdx.x * 256 + threadIdx.x;
  int2 e = e01[t]; float2 w = w01[t];
  int p0 = atomicAdd(&cursor[e.x], 1); token_list[p0] = t; weight_list[p0] = w.x;
  int p1 = atomicAdd(&cursor[e.y], 1); token_list[p1] = t; weight_list[p1] = w.y;
}

// ---------------- 5) gate/up GEMM + SiLU -> h (bf16) ----------------
// grid (MAX_TILES, HDIM/128), block 256. Each wave: one 16-col slice of H,
// TWO 16-row A tiles sharing every B load (4 WMMAs per gate/up B pair).
__global__ __launch_bounds__(256) void gateup_kernel(
    const unsigned short* __restrict__ xbf,
    const unsigned short* __restrict__ wgb, const unsigned short* __restrict__ wub,
    const int* __restrict__ token_list,
    const int* __restrict__ tiles_e, const int* __restrict__ tiles_base,
    const int* __restrict__ ntiles,
    unsigned short* __restrict__ h_buf) {
  __shared__ int toks[MTILE];
  __shared__ __align__(16) unsigned short As[2][MTILE][72]; // 144B row stride, 16B-aligned
  int tile = blockIdx.x;
  if (tile >= *ntiles) return;
  int e = tiles_e[tile];
  int slot0 = tiles_base[tile];
  int tid = threadIdx.x, lane = tid & 31, wv = tid >> 5;
  if (tid < MTILE) toks[tid] = token_list[slot0 + tid];
  __syncthreads();
  int hbase = blockIdx.y * 128 + wv * 16;
  int n = lane & 15, hi = lane >> 4;       // n doubles as A-row m for this lane
  const unsigned short* gRow = wgb + ((size_t)e * HDIM + hbase + n) * DMODEL + hi * 16;
  const unsigned short* uRow = wub + ((size_t)e * HDIM + hbase + n) * DMODEL + hi * 16;
  int arow = tid >> 3, acol = (tid & 7) * 8;             // coop A stage: 16B per thread
  const unsigned short* aPtr = xbf + (size_t)toks[arow] * DMODEL + acol;
  v8f ag0 = {0.f,0.f,0.f,0.f,0.f,0.f,0.f,0.f};
  v8f ag1 = {0.f,0.f,0.f,0.f,0.f,0.f,0.f,0.f};
  v8f au0 = {0.f,0.f,0.f,0.f,0.f,0.f,0.f,0.f};
  v8f au1 = {0.f,0.f,0.f,0.f,0.f,0.f,0.f,0.f};
  v4u areg = *(const v4u*)aPtr;                           // chunk 0
  *(v4u*)&As[0][arow][acol] = areg;
  const int NCH = DMODEL / KCHUNK;                        // 16
  for (int c = 0; c < NCH; ++c) {
    int kb = c * KCHUNK;
    if (c + 1 < NCH) {
      areg = *(const v4u*)(aPtr + kb + KCHUNK);           // hide global latency under compute
      __builtin_prefetch(gRow + kb + KCHUNK, 0, 1);
      __builtin_prefetch(uRow + kb + KCHUNK, 0, 1);
    }
    __syncthreads();                                      // As[c&1] ready; other buffer free
    const unsigned short (*Ab)[72] = As[c & 1];
#pragma unroll
    for (int s = 0; s < KCHUNK / 32; ++s) {
      AVec a0, a1;
      a0.q.lo = *(const v4u*)&Ab[n][s * 32 + hi * 16];          // rows 0..15
      a0.q.hi = *(const v4u*)&Ab[n][s * 32 + hi * 16 + 8];
      a1.q.lo = *(const v4u*)&Ab[16 + n][s * 32 + hi * 16];     // rows 16..31
      a1.q.hi = *(const v4u*)&Ab[16 + n][s * 32 + hi * 16 + 8];
      v16bf bg = *(const v16bf*)(gRow + kb + s * 32);          // shared by both A tiles
      v16bf bu = *(const v16bf*)(uRow + kb + s * 32);
      ag0 = __builtin_amdgcn_wmma_f32_16x16x32_bf16(false, a0.v, false, bg, (short)0, ag0, false, false);
      ag1 = __builtin_amdgcn_wmma_f32_16x16x32_bf16(false, a1.v, false, bg, (short)0, ag1, false, false);
      au0 = __builtin_amdgcn_wmma_f32_16x16x32_bf16(false, a0.v, false, bu, (short)0, au0, false, false);
      au1 = __builtin_amdgcn_wmma_f32_16x16x32_bf16(false, a1.v, false, bu, (short)0, au1, false, false);
    }
    if (c + 1 < NCH) *(v4u*)&As[(c + 1) & 1][arow][acol] = areg;
  }
#pragma unroll
  for (int r = 0; r < 8; ++r) {
    int m = r + 8 * hi;                                   // C layout: M = r + 8*hi
    float g0 = ag0[r], u0 = au0[r];
    float h0 = (g0 / (1.f + __expf(-g0))) * u0;           // SiLU(g) * u
    h_buf[(size_t)(slot0 + m) * HDIM + hbase + n] = (unsigned short)f2bf1(h0);
    float g1 = ag1[r], u1 = au1[r];
    float h1 = (g1 / (1.f + __expf(-g1))) * u1;
    h_buf[(size_t)(slot0 + 16 + m) * HDIM + hbase + n] = (unsigned short)f2bf1(h1);
  }
}

// ---------------- 6) down GEMM + weighted scatter-add ----------------
// grid (MAX_TILES, DMODEL/128), block 256; two A tiles share each B load.
__global__ __launch_bounds__(256) void down_kernel(
    const unsigned short* __restrict__ h_buf,
    const unsigned short* __restrict__ wdb,
    const int* __restrict__ token_list, const float* __restrict__ weight_list,
    const int* __restrict__ tiles_e, const int* __restrict__ tiles_base,
    const int* __restrict__ ntiles,
    float* __restrict__ out) {
  __shared__ __align__(16) unsigned short As[2][MTILE][72];
  int tile = blockIdx.x;
  if (tile >= *ntiles) return;
  int e = tiles_e[tile];
  int slot0 = tiles_base[tile];
  int tid = threadIdx.x, lane = tid & 31, wv = tid >> 5;
  int dbase = blockIdx.y * 128 + wv * 16;
  int n = lane & 15, hi = lane >> 4;
  const unsigned short* dRow = wdb + ((size_t)e * DMODEL + dbase + n) * HDIM + hi * 16;
  int arow = tid >> 3, acol = (tid & 7) * 8;
  const unsigned short* aPtr = h_buf + (size_t)(slot0 + arow) * HDIM + acol;
  v8f ac0 = {0.f,0.f,0.f,0.f,0.f,0.f,0.f,0.f};
  v8f ac1 = {0.f,0.f,0.f,0.f,0.f,0.f,0.f,0.f};
  v4u areg = *(const v4u*)aPtr;
  *(v4u*)&As[0][arow][acol] = areg;
  const int NCH = HDIM / KCHUNK;                          // 32
  for (int c = 0; c < NCH; ++c) {
    int kb = c * KCHUNK;
    if (c + 1 < NCH) {
      areg = *(const v4u*)(aPtr + kb + KCHUNK);
      __builtin_prefetch(dRow + kb + KCHUNK, 0, 1);
    }
    __syncthreads();
    const unsigned short (*Ab)[72] = As[c & 1];
#pragma unroll
    for (int s = 0; s < KCHUNK / 32; ++s) {
      AVec a0, a1;
      a0.q.lo = *(const v4u*)&Ab[n][s * 32 + hi * 16];
      a0.q.hi = *(const v4u*)&Ab[n][s * 32 + hi * 16 + 8];
      a1.q.lo = *(const v4u*)&Ab[16 + n][s * 32 + hi * 16];
      a1.q.hi = *(const v4u*)&Ab[16 + n][s * 32 + hi * 16 + 8];
      v16bf bd = *(const v16bf*)(dRow + kb + s * 32);
      ac0 = __builtin_amdgcn_wmma_f32_16x16x32_bf16(false, a0.v, false, bd, (short)0, ac0, false, false);
      ac1 = __builtin_amdgcn_wmma_f32_16x16x32_bf16(false, a1.v, false, bd, (short)0, ac1, false, false);
    }
    if (c + 1 < NCH) *(v4u*)&As[(c + 1) & 1][arow][acol] = areg;
  }
#pragma unroll
  for (int r = 0; r < 8; ++r) {
    int m = r + 8 * hi;
    int s0 = slot0 + m;
    int s1 = slot0 + 16 + m;
    atomicAdd(out + (size_t)token_list[s0] * DMODEL + dbase + n, ac0[r] * weight_list[s0]);
    atomicAdd(out + (size_t)token_list[s1] * DMODEL + dbase + n, ac1[r] * weight_list[s1]);
  }
}

extern "C" void kernel_launch(void* const* d_in, const int* in_sizes, int n_in,
                              void* d_out, int out_size, void* d_ws, size_t ws_size,
                              hipStream_t stream) {
  (void)in_sizes; (void)n_in; (void)out_size; (void)ws_size;
  const float* x  = (const float*)d_in[0];  // [T, D]
  const float* rw = (const float*)d_in[1];  // [E, D]
  const float* wg = (const float*)d_in[2];  // [E, H, D]
  const float* wu = (const float*)d_in[3];  // [E, H, D]
  const float* wd = (const float*)d_in[4];  // [E, D, H]
  float* out = (float*)d_out;               // [T, D]

  char* base = (char*)d_ws;
  size_t off = 0;
  auto take = [&](size_t bytes) -> char* {
    char* r = base + off;
    off = (off + bytes + 255) & ~(size_t)255;
    return r;
  };
  const size_t WELEM = (size_t)NEXP * HDIM * DMODEL;     // 16,777,216 per weight tensor
  unsigned short* xbf  = (unsigned short*)take((size_t)T_TOK * DMODEL * 2);  // 16 MB
  unsigned short* wgb  = (unsigned short*)take(WELEM * 2);                   // 32 MB
  unsigned short* wub  = (unsigned short*)take(WELEM * 2);                   // 32 MB
  unsigned short* wdb  = (unsigned short*)take(WELEM * 2);                   // 32 MB
  unsigned short* hbuf = (unsigned short*)take((size_t)NSLOT * HDIM * 2);    // 65 MB
  int*    tok    = (int*)   take((size_t)NSLOT * 4);
  float*  wlist  = (float*) take((size_t)NSLOT * 4);
  int2*   e01    = (int2*)  take((size_t)T_TOK * 8);
  float2* w01    = (float2*)take((size_t)T_TOK * 8);
  int*    counts = (int*)   take(64);
  int*    cursor = (int*)   take(64);
  int*    te     = (int*)   take((size_t)MAX_TILES * 4);
  int*    tb     = (int*)   take((size_t)MAX_TILES * 4);
  int*    ntl    = (int*)   take(64);

  hipMemsetAsync(counts, 0, NEXP * sizeof(int), stream);
  hipMemsetAsync(out, 0, (size_t)T_TOK * DMODEL * sizeof(float), stream);

  // Pre-pass casts (bandwidth bound; removes all cvt VALU from GEMM loops)
  cast_bf16_kernel<<<(T_TOK * DMODEL) / (4 * 256), 256, 0, stream>>>(x, xbf);
  cast_bf16_kernel<<<WELEM / (4 * 256), 256, 0, stream>>>(wg, wgb);
  cast_bf16_kernel<<<WELEM / (4 * 256), 256, 0, stream>>>(wu, wub);
  cast_bf16_kernel<<<WELEM / (4 * 256), 256, 0, stream>>>(wd, wdb);

  router_kernel<<<T_TOK / 8, 256, 0, stream>>>(x, rw, e01, w01, counts);
  setup_kernel<<<1, 32, 0, stream>>>(counts, cursor, tok, wlist, te, tb, ntl);
  scatter_kernel<<<T_TOK / 256, 256, 0, stream>>>(e01, w01, cursor, tok, wlist);
  gateup_kernel<<<dim3(MAX_TILES, HDIM / 128), 256, 0, stream>>>(xbf, wgb, wub, tok, te, tb, ntl, hbuf);
  down_kernel<<<dim3(MAX_TILES, DMODEL / 128), 256, 0, stream>>>(hbuf, wdb, tok, wlist, te, tb, ntl, out);
}
